// SLSTMBlock_87686052315888
// MI455X (gfx1250) — matrix operand order repaired and checked by reading, not compiled, not run on hardware
//
#include <hip/hip_runtime.h>

// ---------------------------------------------------------------------------
// Problem constants
// ---------------------------------------------------------------------------
#define BB   4
#define TT   2048
#define HH   8
#define KK   128
#define DD   1024          // H*K
#define MM   8192          // B*T
#define F1   1365
#define F1P  1408          // padded to 64-multiple for WMMA tiling
#define EPSF 1e-6f

// LDS strides (elements). 40*2 = 80 bytes = multiple of 16 -> b128-aligned rows.
#define LDA  40
#define LDB  40

// ---------------------------------------------------------------------------
// bf16 helpers (manual RNE conversion; storage as unsigned short)
// ---------------------------------------------------------------------------
typedef __attribute__((ext_vector_type(16))) __bf16 v16bf;
typedef __attribute__((ext_vector_type(8)))  float  v8f;

struct B32x8 { uint4 lo, hi; };   // 32 bytes == one WMMA 16-bit fragment

static __device__ __forceinline__ unsigned short f2bf(float f) {
  union { float f; unsigned u; } v; v.f = f;
  unsigned r = v.u + 0x7FFFu + ((v.u >> 16) & 1u);
  return (unsigned short)(r >> 16);
}
static __device__ __forceinline__ float bf2f(unsigned short h) {
  union { unsigned u; float f; } v; v.u = ((unsigned)h) << 16;
  return v.f;
}
static __device__ __forceinline__ float sigmoidf_(float x) {
  return 1.0f / (1.0f + __expf(-x));
}

// CDNA5 async tile fill: GLOBAL_LOAD_ASYNC_TO_LDS_B128 (ASYNCcnt-tracked,
// no VGPR return data). LDS address operand = flat address truncated to
// [31:0] (ISA 10.2 LDS aperture rule).
static __device__ __forceinline__ void async_b128(void* lds_dst, const void* gsrc) {
  unsigned lds = (unsigned)(uintptr_t)lds_dst;
  unsigned long long ga = (unsigned long long)(uintptr_t)gsrc;
  asm volatile("global_load_async_to_lds_b128 %0, %1, off"
               :: "v"(lds), "v"(ga) : "memory");
}
static __device__ __forceinline__ void async_wait0() {
  asm volatile("s_wait_asynccnt 0x0" ::: "memory");
}

// ---------------------------------------------------------------------------
// WMMA bf16 16x16x32 fragment builders (ISA 7.12.2 layouts, wave32)
// A (16x32), LDS row-major tile: two aligned ds_load_b128 per lane
// B (32x16), LDS tile stored TRANSPOSED (n-major): two ds_load_b128 per lane
// C/D (16x16 f32): VGPR i -> M = i + hi*8, N = lane&15
// ---------------------------------------------------------------------------
static __device__ __forceinline__ v16bf frag_a(const unsigned short* As, int lane) {
  const unsigned short* p = As + (lane & 15) * LDA + ((lane >> 4) ? 8 : 0);
  B32x8 t;
  t.lo = *(const uint4*)(p);
  t.hi = *(const uint4*)(p + 16);
  return __builtin_bit_cast(v16bf, t);
}
static __device__ __forceinline__ v16bf frag_bT(const unsigned short* BsT, int lane) {
  const unsigned short* p = BsT + (lane & 15) * LDB + (lane >> 4) * 8;
  B32x8 t;
  t.lo = *(const uint4*)(p);
  t.hi = *(const uint4*)(p + 16);
  return __builtin_bit_cast(v16bf, t);
}
static __device__ __forceinline__ v8f wmma_bf16(v16bf a, v16bf b, v8f c) {
  return __builtin_amdgcn_wmma_f32_16x16x32_bf16(
      false, a, false, b, (short)0, c, false, false);
}

// ---------------------------------------------------------------------------
// Weight prep: f32 -> bf16, TRANSPOSED so GEMM B tiles are n-major, padded.
// ---------------------------------------------------------------------------
// conv: src (w, i, o) -> dst [w][o][i]
__global__ void cvt_convT(const float* __restrict__ src,
                          unsigned short* __restrict__ dst) {
  const int n = 4 * DD * DD;
  for (int i = blockIdx.x * blockDim.x + threadIdx.x; i < n;
       i += gridDim.x * blockDim.x) {
    int w   = i / (DD * DD);
    int rem = i - w * DD * DD;
    int o   = rem >> 10;          // out channel (row of dst)
    int ii  = rem & 1023;         // in channel  (col of dst)
    dst[i] = f2bf(src[(w * DD + ii) * DD + o]);
  }
}

// gates: src_g (j, h, k) -> dst [gate][head][k][j]
__global__ void cvt_gatesT(const float* __restrict__ wz,
                           const float* __restrict__ wi,
                           const float* __restrict__ wf,
                           const float* __restrict__ wo,
                           unsigned short* __restrict__ dst) {
  const int n = 4 * HH * KK * KK;
  for (int i = blockIdx.x * blockDim.x + threadIdx.x; i < n;
       i += gridDim.x * blockDim.x) {
    int gate = i >> 17;
    int rem  = i & 131071;
    int head = rem >> 14;
    int kj   = rem & 16383;
    int k    = kj >> 7;           // output index (row of dst)
    int j    = kj & 127;          // input index  (col of dst)
    const float* w = (gate == 0) ? wz : (gate == 1) ? wi : (gate == 2) ? wf : wo;
    dst[i] = f2bf(w[(j * HH + head) * KK + k]);
  }
}

// transpose + pad: logical source S is sr x sc (row-major); dst = S^T padded
// to dr x dc: dst[r*dc + c] = (r < sc && c < sr) ? S[c*sc + r] : 0
__global__ void cvt_padT(const float* __restrict__ src,
                         unsigned short* __restrict__ dst,
                         int sr, int sc, int dr, int dc) {
  const int n = dr * dc;
  for (int i = blockIdx.x * blockDim.x + threadIdx.x; i < n;
       i += gridDim.x * blockDim.x) {
    int r = i / dc, c = i - r * dc;
    dst[i] = (r < sc && c < sr) ? f2bf(src[c * sc + r]) : (unsigned short)0;
  }
}

// ---------------------------------------------------------------------------
// LayerNorm over D=1024, one block per token row -> bf16 xn
// ---------------------------------------------------------------------------
__global__ void layernorm_k(const float* __restrict__ x,
                            const float* __restrict__ scale,
                            const float* __restrict__ bias,
                            unsigned short* __restrict__ xn) {
  __shared__ float r1[256], r2[256];
  const int row = blockIdx.x, tid = threadIdx.x;
  float v[4], s = 0.f, sq = 0.f;
#pragma unroll
  for (int i = 0; i < 4; ++i) {
    v[i] = x[row * DD + tid + i * 256];
    s += v[i]; sq += v[i] * v[i];
  }
  r1[tid] = s; r2[tid] = sq;
  __syncthreads();
  for (int o = 128; o > 0; o >>= 1) {
    if (tid < o) { r1[tid] += r1[tid + o]; r2[tid] += r2[tid + o]; }
    __syncthreads();
  }
  const float mean = r1[0] * (1.0f / DD);
  const float var  = r2[0] * (1.0f / DD) - mean * mean;
  const float rstd = rsqrtf(var + EPSF);
#pragma unroll
  for (int i = 0; i < 4; ++i) {
    int d = tid + i * 256;
    xn[row * DD + d] = f2bf((v[i] - mean) * rstd * scale[d] + bias[d]);
  }
}

// ---------------------------------------------------------------------------
// Conv1d(k=4, SAME) as 4 accumulated shifted GEMMs + SiLU -> i_f (bf16)
// Workgroup tile 32(M) x 128(N); 8 waves (2x4); each wave: 2 accumulators.
// B tiles filled via async-DMA to LDS; A tile synchronous (needs boundary
// predicate) and overlaps with the async stream.
// ---------------------------------------------------------------------------
__global__ void conv_silu_gemm(const unsigned short* __restrict__ xn,
                               const unsigned short* __restrict__ wcT,
                               const float* __restrict__ conv_b,
                               unsigned short* __restrict__ out) {
  __shared__ unsigned short As[32][LDA];      // 32 x 32 used
  __shared__ unsigned short BsT[128][LDB];    // n-major, 128 x 32 used
  const int tid = threadIdx.x, lane = tid & 31, wave = tid >> 5;
  const int wm = wave >> 2, wn = wave & 3;
  const int m0 = blockIdx.x * 32, n0 = blockIdx.y * 128;
  v8f acc0 = {}, acc1 = {};
  for (int w = 0; w < 4; ++w) {
    for (int kk = 0; kk < DD; kk += 32) {
      // B tile: 512 x b128 chunks (2/thread) via async-to-LDS
#pragma unroll
      for (int i = 0; i < 2; ++i) {
        int c = tid + i * 256;                // 0..511
        int n = c >> 2, ko = (c & 3) * 8;
        async_b128(&BsT[n][ko],
                   &wcT[(size_t)(w * DD + n0 + n) * DD + kk + ko]);
      }
      // A tile: 256 x b64 chunks (1/thread); shift/boundary uniform per row
      {
        int m = tid >> 3, ko = (tid & 7) * 4;
        int r = m0 + m;
        int ts = (r & (TT - 1)) + w - 1;
        uint2 val = make_uint2(0u, 0u);
        if (ts >= 0 && ts < TT)
          val = *(const uint2*)&xn[(r + w - 1) * DD + kk + ko];
        *(uint2*)&As[m][ko] = val;
      }
      if (kk + 32 < DD)   // global_prefetch hint for next weight chunk
        __builtin_prefetch(&wcT[(size_t)(w * DD + n0 + (tid >> 1)) * DD + kk + 32], 0, 1);
      async_wait0();
      __syncthreads();
      v16bf a  = frag_a(&As[wm * 16][0], lane);
      v16bf b0 = frag_bT(&BsT[wn * 16][0], lane);
      acc0 = wmma_bf16(a, b0, acc0);
      v16bf b1 = frag_bT(&BsT[wn * 16 + 64][0], lane);
      acc1 = wmma_bf16(a, b1, acc1);
      __syncthreads();
    }
  }
  const int hi = lane >> 4, col = lane & 15;
  const int na = n0 + wn * 16 + col, nb = na + 64;
  const float ba = conv_b[na], bb = conv_b[nb];
#pragma unroll
  for (int i = 0; i < 8; ++i) {
    int m = m0 + wm * 16 + hi * 8 + i;
    float xa = acc0[i] + ba;
    float xb = acc1[i] + bb;
    out[m * DD + na] = f2bf(xa * sigmoidf_(xa));
    out[m * DD + nb] = f2bf(xb * sigmoidf_(xb));
  }
}

// ---------------------------------------------------------------------------
// Per-head gate projections: tile 32(M) x 128(N = full head), K = 128.
// blockIdx.y = gate*8 + head. gate 0=z (A=xn), 1=i, 2=f (A=i_f), 3=o (A=xn)
// ---------------------------------------------------------------------------
__global__ void proj_gemm(const unsigned short* __restrict__ xn,
                          const unsigned short* __restrict__ ifb,
                          const unsigned short* __restrict__ wgT,
                          const float* __restrict__ bz,
                          const float* __restrict__ bi,
                          const float* __restrict__ bff,
                          const float* __restrict__ bo,
                          unsigned short* __restrict__ sz,
                          unsigned short* __restrict__ si,
                          unsigned short* __restrict__ sf,
                          unsigned short* __restrict__ so) {
  __shared__ unsigned short As[32][LDA];
  __shared__ unsigned short BsT[128][LDB];
  const int tid = threadIdx.x, lane = tid & 31, wave = tid >> 5;
  const int wm = wave >> 2, wn = wave & 3;
  const int gate = blockIdx.y >> 3, head = blockIdx.y & 7;
  const int m0 = blockIdx.x * 32;
  const unsigned short* A = (gate == 1 || gate == 2) ? ifb : xn;
  const float* bias = (gate == 0) ? bz : (gate == 1) ? bi : (gate == 2) ? bff : bo;
  unsigned short* out = (gate == 0) ? sz : (gate == 1) ? si : (gate == 2) ? sf : so;
  const unsigned short* WhT = wgT + (size_t)(gate * HH + head) * KK * KK;
  v8f acc0 = {}, acc1 = {};
  for (int kk = 0; kk < KK; kk += 32) {
#pragma unroll
    for (int i = 0; i < 2; ++i) {
      int c = tid + i * 256;
      int n = c >> 2, ko = (c & 3) * 8;
      async_b128(&BsT[n][ko], &WhT[n * KK + kk + ko]);
    }
    {
      int m = tid >> 3, ko = (tid & 7) * 4;
      *(uint2*)&As[m][ko] =
          *(const uint2*)&A[(m0 + m) * DD + head * KK + kk + ko];
    }
    async_wait0();
    __syncthreads();
    v16bf a  = frag_a(&As[wm * 16][0], lane);
    v16bf b0 = frag_bT(&BsT[wn * 16][0], lane);
    acc0 = wmma_bf16(a, b0, acc0);
    v16bf b1 = frag_bT(&BsT[wn * 16 + 64][0], lane);
    acc1 = wmma_bf16(a, b1, acc1);
    __syncthreads();
  }
  const int hi = lane >> 4, col = lane & 15;
  const int na = wn * 16 + col, nb = na + 64;      // within-head (0..127)
  const float ba = bias[na], bb = bias[nb];
#pragma unroll
  for (int i = 0; i < 8; ++i) {
    int m = m0 + wm * 16 + hi * 8 + i;
    out[m * DD + head * KK + na] = f2bf(acc0[i] + ba);
    out[m * DD + head * KK + nb] = f2bf(acc1[i] + bb);
  }
}

// ---------------------------------------------------------------------------
// sLSTM sequential scan. One workgroup per (b, h); 512 threads = (gate, k).
// Recurrent weights transposed into dynamic LDS: Rt[4][128][133] f32 (~266KB,
// needs CDNA5's 320KB/WGP LDS). GroupNorm stats fused in-flight.
// ---------------------------------------------------------------------------
__global__ void slstm_scan(const unsigned short* __restrict__ sz,
                           const unsigned short* __restrict__ si,
                           const unsigned short* __restrict__ sf,
                           const unsigned short* __restrict__ so,
                           const float* __restrict__ rz,
                           const float* __restrict__ ri,
                           const float* __restrict__ rf_,
                           const float* __restrict__ ro,
                           unsigned short* __restrict__ hout,
                           float* __restrict__ gstat) {
  extern __shared__ float sm[];
  float* Rt   = sm;                       // [4][128][133]
  float* pre  = sm + 4 * 128 * 133;       // [4][128]
  float* hld  = pre + 512;                // [128]
  float* red1 = hld + 128;                // [128]
  float* red2 = red1 + 128;               // [128]

  const int b = blockIdx.x >> 3, hh = blockIdx.x & 7;
  const int tid = threadIdx.x;
  const int g = tid >> 7, k = tid & 127;

  const float* R = (g == 0) ? rz : (g == 1) ? ri : (g == 2) ? rf_ : ro;
  {
    float* dst = Rt + (size_t)(g * 128 + k) * 133;
#pragma unroll 4
    for (int j = 0; j < 128; ++j)
      dst[j] = R[(j * HH + hh) * KK + k];   // Rt[g][k][j] = r[j, h, k]
  }
  if (g == 0) hld[k] = 0.f;
  __syncthreads();

  const unsigned short* sg = (g == 0) ? sz : (g == 1) ? si : (g == 2) ? sf : so;
  const float* myR = Rt + (size_t)(g * 128 + k) * 133;
  float c = 0.f, nacc = 0.f, s1 = 0.f, s2 = 0.f;
  int base = (b * TT) * DD + hh * KK + k;

  for (int t = 0; t < TT; ++t) {
    float acc = bf2f(sg[base]);
#pragma unroll 8
    for (int j = 0; j < 128; ++j) acc += hld[j] * myR[j];
    pre[g * 128 + k] = acc;
    __syncthreads();
    if (g == 0) {
      float zt = tanhf(pre[k]);
      float it = __expf(pre[128 + k]);
      float ft = __expf(pre[256 + k]);
      float ot = sigmoidf_(pre[384 + k]);
      c    = ft * c + it * zt;
      nacc = ft * nacc + it;
      float hv = ot * (c / nacc);
      hout[base] = f2bf(hv);
      hld[k] = hv;
      s1 += hv; s2 += hv * hv;
    }
    __syncthreads();
    base += DD;
  }
  if (g == 0) { red1[k] = s1; red2[k] = s2; }
  __syncthreads();
  if (tid == 0) {
    float S = 0.f, Q = 0.f;
    for (int i = 0; i < 128; ++i) { S += red1[i]; Q += red2[i]; }
    const float inv = 1.0f / (float)(TT * KK);
    float mean = S * inv;
    float var  = Q * inv - mean * mean;
    gstat[(b * HH + hh) * 2]     = mean;
    gstat[(b * HH + hh) * 2 + 1] = rsqrtf(var + EPSF);
  }
}

// ---------------------------------------------------------------------------
// GroupNorm apply + affine -> g (bf16)
// ---------------------------------------------------------------------------
__global__ void gn_apply(const unsigned short* __restrict__ h,
                         const float* __restrict__ gstat,
                         const float* __restrict__ gsc,
                         const float* __restrict__ gbi,
                         unsigned short* __restrict__ g) {
  const int n = MM * DD;
  for (int i = blockIdx.x * blockDim.x + threadIdx.x; i < n;
       i += gridDim.x * blockDim.x) {
    int row = i >> 10, d = i & 1023;
    int b = row >> 11, hd = d >> 7;
    float mean = gstat[(b * HH + hd) * 2];
    float rstd = gstat[(b * HH + hd) * 2 + 1];
    g[i] = f2bf((bf2f(h[i]) - mean) * rstd * gsc[d] + gbi[d]);
  }
}

// ---------------------------------------------------------------------------
// MLP stage 1: dual GEMM sharing the A fragment:
//   p = (g@w1 + b1) * silu(g@w2 + b2)      [M x F1P], bf16
// ---------------------------------------------------------------------------
__global__ void mlp1_gemm(const unsigned short* __restrict__ g,
                          const unsigned short* __restrict__ w1T,
                          const unsigned short* __restrict__ w2T,
                          const float* __restrict__ b1,
                          const float* __restrict__ b2,
                          unsigned short* __restrict__ p) {
  __shared__ unsigned short As[32][LDA];
  __shared__ unsigned short B1T[64][LDB];
  __shared__ unsigned short B2T[64][LDB];
  const int tid = threadIdx.x, lane = tid & 31, wave = tid >> 5;
  const int wm = wave >> 2, wn = wave & 3;
  const int m0 = blockIdx.x * 32, n0 = blockIdx.y * 64;
  v8f acc1 = {}, acc2 = {};
  for (int kk = 0; kk < DD; kk += 32) {
    {
      int n = tid >> 2, ko = (tid & 3) * 8;   // 256 x b128 chunks, 1/thread each
      async_b128(&B1T[n][ko], &w1T[(size_t)(n0 + n) * DD + kk + ko]);
      async_b128(&B2T[n][ko], &w2T[(size_t)(n0 + n) * DD + kk + ko]);
    }
    {
      int m = tid >> 3, ko = (tid & 7) * 4;
      *(uint2*)&As[m][ko] = *(const uint2*)&g[(m0 + m) * DD + kk + ko];
    }
    async_wait0();
    __syncthreads();
    v16bf a  = frag_a(&As[wm * 16][0], lane);
    v16bf f1 = frag_bT(&B1T[wn * 16][0], lane);
    acc1 = wmma_bf16(a, f1, acc1);
    v16bf f2 = frag_bT(&B2T[wn * 16][0], lane);
    acc2 = wmma_bf16(a, f2, acc2);
    __syncthreads();
  }
  const int hi = lane >> 4, col = lane & 15;
  const int n = n0 + wn * 16 + col;
  const float bb1 = (n < F1) ? b1[n] : 0.f;
  const float bb2 = (n < F1) ? b2[n] : 0.f;
#pragma unroll
  for (int i = 0; i < 8; ++i) {
    int m = m0 + wm * 16 + hi * 8 + i;
    float x1 = acc1[i] + bb1;
    float x2 = acc2[i] + bb2;
    p[(size_t)m * F1P + n] = f2bf(x1 * (x2 * sigmoidf_(x2)));
  }
}

// ---------------------------------------------------------------------------
// MLP stage 2: y = p @ w3 + b3 + skip  (f32 output)
// tile 32(M) x 128(N), K = F1P, async B-tile fill
// ---------------------------------------------------------------------------
__global__ void mlp2_gemm(const unsigned short* __restrict__ p,
                          const unsigned short* __restrict__ w3T,
                          const float* __restrict__ b3,
                          const float* __restrict__ skip,
                          float* __restrict__ y) {
  __shared__ unsigned short As[32][LDA];
  __shared__ unsigned short BsT[128][LDB];
  const int tid = threadIdx.x, lane = tid & 31, wave = tid >> 5;
  const int wm = wave >> 2, wn = wave & 3;
  const int m0 = blockIdx.x * 32, n0 = blockIdx.y * 128;
  v8f acc0 = {}, acc1 = {};
  for (int kk = 0; kk < F1P; kk += 32) {
#pragma unroll
    for (int i = 0; i < 2; ++i) {
      int c = tid + i * 256;
      int n = c >> 2, ko = (c & 3) * 8;
      async_b128(&BsT[n][ko], &w3T[(size_t)(n0 + n) * F1P + kk + ko]);
    }
    {
      int m = tid >> 3, ko = (tid & 7) * 4;
      *(uint2*)&As[m][ko] =
          *(const uint2*)&p[(size_t)(m0 + m) * F1P + kk + ko];
    }
    async_wait0();
    __syncthreads();
    v16bf a  = frag_a(&As[wm * 16][0], lane);
    v16bf b0 = frag_bT(&BsT[wn * 16][0], lane);
    acc0 = wmma_bf16(a, b0, acc0);
    v16bf b1 = frag_bT(&BsT[wn * 16 + 64][0], lane);
    acc1 = wmma_bf16(a, b1, acc1);
    __syncthreads();
  }
  const int hi = lane >> 4, col = lane & 15;
  const int na = n0 + wn * 16 + col, nb = na + 64;
  const float ba = b3[na], bbv = b3[nb];
#pragma unroll
  for (int i = 0; i < 8; ++i) {
    int m = m0 + wm * 16 + hi * 8 + i;
    y[m * DD + na] = acc0[i] + ba  + skip[m * DD + na];
    y[m * DD + nb] = acc1[i] + bbv + skip[m * DD + nb];
  }
}

// ---------------------------------------------------------------------------
// Workspace layout (bytes)
// ---------------------------------------------------------------------------
static constexpr size_t OFF_WC    = 0;                                   // wcT [4][DD][DD]
static constexpr size_t OFF_WG    = OFF_WC    + (size_t)4 * DD * DD * 2; // wgT [4][H][K][K]
static constexpr size_t OFF_W1P   = OFF_WG    + (size_t)4 * HH * KK * KK * 2; // w1T [F1P][DD]
static constexpr size_t OFF_W2P   = OFF_W1P   + (size_t)F1P * DD * 2;
static constexpr size_t OFF_W3P   = OFF_W2P   + (size_t)F1P * DD * 2;    // w3T [DD][F1P]
static constexpr size_t OFF_XN    = OFF_W3P   + (size_t)DD * F1P * 2;
static constexpr size_t OFF_IF    = OFF_XN    + (size_t)MM * DD * 2;
static constexpr size_t OFF_SZ    = OFF_IF    + (size_t)MM * DD * 2;
static constexpr size_t OFF_SI    = OFF_SZ    + (size_t)MM * DD * 2;
static constexpr size_t OFF_SF    = OFF_SI    + (size_t)MM * DD * 2;
static constexpr size_t OFF_SO    = OFF_SF    + (size_t)MM * DD * 2;
static constexpr size_t OFF_H     = OFF_SO    + (size_t)MM * DD * 2;
static constexpr size_t OFF_G     = OFF_H     + (size_t)MM * DD * 2;
static constexpr size_t OFF_P     = OFF_G     + (size_t)MM * DD * 2;
static constexpr size_t OFF_GSTAT = OFF_P     + (size_t)MM * F1P * 2;

extern "C" void kernel_launch(void* const* d_in, const int* in_sizes, int n_in,
                              void* d_out, int out_size, void* d_ws, size_t ws_size,
                              hipStream_t stream) {
  const float* x        = (const float*)d_in[0];
  const float* ln_scale = (const float*)d_in[1];
  const float* ln_bias  = (const float*)d_in[2];
  const float* conv_k   = (const float*)d_in[3];
  const float* conv_b   = (const float*)d_in[4];
  const float* wz       = (const float*)d_in[5];
  const float* wi       = (const float*)d_in[6];
  const float* wf       = (const float*)d_in[7];
  const float* wo       = (const float*)d_in[8];
  const float* bz       = (const float*)d_in[9];
  const float* bi       = (const float*)d_in[10];
  const float* bf       = (const float*)d_in[11];
  const float* bo       = (const float*)d_in[12];
  const float* rz       = (const float*)d_in[13];
  const float* ri       = (const float*)d_in[14];
  const float* rf       = (const float*)d_in[15];
  const float* ro       = (const float*)d_in[16];
  const float* gn_scale = (const float*)d_in[17];
  const float* gn_bias  = (const float*)d_in[18];
  const float* w1       = (const float*)d_in[19];
  const float* b1       = (const float*)d_in[20];
  const float* w2       = (const float*)d_in[21];
  const float* b2       = (const float*)d_in[22];
  const float* w3       = (const float*)d_in[23];
  const float* b3       = (const float*)d_in[24];
  float* y = (float*)d_out;

  char* ws = (char*)d_ws;
  unsigned short* wcT  = (unsigned short*)(ws + OFF_WC);
  unsigned short* wgT  = (unsigned short*)(ws + OFF_WG);
  unsigned short* w1T  = (unsigned short*)(ws + OFF_W1P);
  unsigned short* w2T  = (unsigned short*)(ws + OFF_W2P);
  unsigned short* w3T  = (unsigned short*)(ws + OFF_W3P);
  unsigned short* xnb  = (unsigned short*)(ws + OFF_XN);
  unsigned short* ifb  = (unsigned short*)(ws + OFF_IF);
  unsigned short* szb  = (unsigned short*)(ws + OFF_SZ);
  unsigned short* sib  = (unsigned short*)(ws + OFF_SI);
  unsigned short* sfb  = (unsigned short*)(ws + OFF_SF);
  unsigned short* sob  = (unsigned short*)(ws + OFF_SO);
  unsigned short* hb   = (unsigned short*)(ws + OFF_H);
  unsigned short* gb   = (unsigned short*)(ws + OFF_G);
  unsigned short* pb   = (unsigned short*)(ws + OFF_P);
  float*          gst  = (float*)(ws + OFF_GSTAT);

  // ---- weight prep (transposed bf16) ----
  cvt_convT <<<2048, 256, 0, stream>>>(conv_k, wcT);
  cvt_gatesT<<<512,  256, 0, stream>>>(wz, wi, wf, wo, wgT);
  cvt_padT  <<<1024, 256, 0, stream>>>(w1, w1T, DD, F1, F1P, DD);
  cvt_padT  <<<1024, 256, 0, stream>>>(w2, w2T, DD, F1, F1P, DD);
  cvt_padT  <<<1024, 256, 0, stream>>>(w3, w3T, F1, DD, DD, F1P);

  // ---- LayerNorm ----
  layernorm_k<<<MM, 256, 0, stream>>>(x, ln_scale, ln_bias, xnb);

  // ---- Conv + SiLU (WMMA, async B tiles) ----
  conv_silu_gemm<<<dim3(MM / 32, DD / 128), 256, 0, stream>>>(xnb, wcT, conv_b, ifb);

  // ---- Gate projections (WMMA, per-head block-diagonal, async B tiles) ----
  proj_gemm<<<dim3(MM / 32, 32), 256, 0, stream>>>(
      xnb, ifb, wgT, bz, bi, bf, bo, szb, sib, sfb, sob);

  // ---- Sequential sLSTM scan, fused GroupNorm stats ----
  {
    const size_t smem = (size_t)(4 * 128 * 133 + 512 + 128 + 256) * sizeof(float);
    slstm_scan<<<BB * HH, 512, smem, stream>>>(szb, sib, sfb, sob,
                                               rz, ri, rf, ro, hb, gst);
  }

  // ---- GroupNorm apply ----
  gn_apply<<<4096, 256, 0, stream>>>(hb, gst, gn_scale, gn_bias, gb);

  // ---- GLU MLP (WMMA, async B tiles) ----
  mlp1_gemm<<<dim3(MM / 32, F1P / 64), 256, 0, stream>>>(gb, w1T, w2T, b1, b2, pb);
  mlp2_gemm<<<dim3(MM / 32, DD / 128), 256, 0, stream>>>(pb, w3T, b3, x, y);
}
